// CGLayer_19035295055891
// MI455X (gfx1250) — matrix operand order repaired and checked by reading, not compiled, not run on hardware
//
#include <hip/hip_runtime.h>
#include <cmath>

#define NNODES 256
#define TAU 8

typedef float    v4f  __attribute__((ext_vector_type(4)));
typedef float    v8f  __attribute__((ext_vector_type(8)));
typedef _Float16 v16h __attribute__((ext_vector_type(16)));

// ---------------------------------------------------------------------------
// CG chunk tables: 15 chunks, ordered L-major then (l1,l2) in reference loop
// order (_cgproduct: for L: for l1: for l2: if |l1-l2|<=L<=l1+l2).
//  L=0: (0,0) (1,1) (2,2)
//  L=1: (0,1) (1,0) (1,1) (1,2) (2,1) (2,2)
//  L=2: (0,2) (1,1) (1,2) (2,0) (2,1) (2,2)
// ---------------------------------------------------------------------------
constexpr int CH_L1[15] = {0,1,2, 0,1,1,1,2,2, 0,1,1,2,2,2};
constexpr int CH_L2[15] = {0,1,2, 1,0,1,2,1,2, 2,1,2,0,1,2};
constexpr int CH_L [15] = {0,0,0, 1,1,1,1,1,1, 2,2,2,2,2,2};
// offsets into flat CG table; chunk size = (2l1+1)(2l2+1)(2L+1)
constexpr int CH_OFF[15]= {0,1,10, 35,44,53,80,125,170, 245,270,315,390,415,490};
constexpr int CH_K [15] = {0,1,2, 0,1,2,3,4,5, 0,1,2,3,4,5}; // chunk idx within its L
constexpr int CG_TOTAL  = 615;
constexpr int ROWB[3]   = {0,1,4};   // row base of part l in the 9-row (m) layout
constexpr int LBASE[3]  = {0,8,32};  // column base of part l in the 72/80-wide vmp
constexpr int CDIM[3]   = {24,48,48};// wr channel dims per L

// ---------------------------------------------------------------------------
// Device Clebsch-Gordan coefficient (f64 internally, exact enough)
// ---------------------------------------------------------------------------
__device__ inline double dfact(int n) {
    const double f[8] = {1.,1.,2.,6.,24.,120.,720.,5040.};
    return f[n];
}

__device__ float cg_coeff(int j1,int m1,int j2,int m2,int J,int M) {
    if (m1 + m2 != M || J < abs(j1 - j2) || J > j1 + j2) return 0.f;
    double pre = sqrt((2.0*J + 1.0) * dfact(J+j1-j2) * dfact(J-j1+j2) *
                      dfact(j1+j2-J) / dfact(j1+j2+J+1));
    pre *= sqrt(dfact(J+M)*dfact(J-M)*dfact(j1-m1)*dfact(j1+m1)*dfact(j2-m2)*dfact(j2+m2));
    int kmin = max(0, max(j2 - J - m1, j1 + m2 - J));
    int kmax = min(j1 + j2 - J, min(j1 - m1, j2 + m2));
    double s = 0.0;
    for (int k = kmin; k <= kmax; ++k) {
        double den = dfact(k)*dfact(j1+j2-J-k)*dfact(j1-m1-k)*dfact(j2+m2-k)*
                     dfact(J-j2+m1+k)*dfact(J-j1-m2+k);
        s += ((k & 1) ? -1.0 : 1.0) / den;
    }
    return (float)(pre * s);
}

// ---------------------------------------------------------------------------
// Kernel 0: fill the 615-entry CG value table in workspace
// ---------------------------------------------------------------------------
__global__ void cg_init_kernel(float* __restrict__ cg) {
    int t = threadIdx.x;
    for (int ci = 0; ci < 15; ++ci) {
        int l1 = CH_L1[ci], l2 = CH_L2[ci], L = CH_L[ci];
        int n2 = 2*l2 + 1, nM = 2*L + 1;
        int sz = (2*l1 + 1) * n2 * nM;
        for (int i = t; i < sz; i += blockDim.x) {
            int M = i % nM;
            int r = i / nM;
            int n = r % n2;
            int m = r / n2;
            cg[CH_OFF[ci] + i] = cg_coeff(l1, m - l1, l2, n - l2, L, M - L);
        }
    }
}

// ---------------------------------------------------------------------------
// Kernel 1a: adj f32 -> f16
// ---------------------------------------------------------------------------
__global__ void pack_adj_kernel(const float* __restrict__ adj,
                                _Float16* __restrict__ adjh) {
    int tid = blockIdx.x * blockDim.x + threadIdx.x;
    if (tid < NNODES * NNODES) adjh[tid] = (_Float16)adj[tid];
}

// ---------------------------------------------------------------------------
// Kernel 1b: pack v0|v1|v2 into f16 Vpad[256][80] (cols 0..71 real, rest 0)
// ---------------------------------------------------------------------------
__global__ void pack_v_kernel(const float* __restrict__ v0,
                              const float* __restrict__ v1,
                              const float* __restrict__ v2,
                              _Float16* __restrict__ vh) {
    int tid = blockIdx.x * blockDim.x + threadIdx.x;
    if (tid >= NNODES * 80) return;
    int j = tid / 80, col = tid % 80;
    float val = 0.f;
    if (col < 8)        val = v0[j * 8  + col];
    else if (col < 32)  val = v1[j * 24 + (col - 8)];
    else if (col < 72)  val = v2[j * 40 + (col - 32)];
    vh[tid] = (_Float16)val;
}

// ---------------------------------------------------------------------------
// Kernel 2: vmp[256][80] = adj x Vpad via v_wmma_f32_16x16x32_f16 (codegen-
// confirmed builtin). One wave per 16x16 tile, K loop = 8 WMMAs of K=32.
//  A (16x32 f16, v16h): lane m=lane&15, h=lane>>4; elem e -> K = (e>>3)*16
//    + h*8 + (e&7)                      (ISA 7.12.2, 16-bit A 16x32 layout)
//  B (32x16 f16, v16h): lane n=lane&15, h=lane>>4; elem e -> K = (e>>3)*16 + h*8 + (e&7)
//  D (16x16 f32, v8f):  VGPR r = row r (lanes 0-15) / row r+8 (lanes 16-31)
// ---------------------------------------------------------------------------
__global__ void __launch_bounds__(32)
adj_gemm_wmma(const _Float16* __restrict__ adjh,
              const _Float16* __restrict__ vh,
              float* __restrict__ vmp) {
    const int lane = threadIdx.x;
    const int rowB = blockIdx.y * 16;
    const int colB = blockIdx.x * 16;
    const int mlo  = lane & 15;
    const int h    = lane >> 4;

    v8f c = {0.f,0.f,0.f,0.f,0.f,0.f,0.f,0.f};
    for (int k0 = 0; k0 < NNODES; k0 += 32) {
        v16h a, b;
#pragma unroll
        for (int e = 0; e < 16; ++e) {
            const int Ka = (e >> 3) * 16 + h * 8 + (e & 7);
            a[e] = adjh[(rowB + mlo) * NNODES + k0 + Ka];
        }
#pragma unroll
        for (int e = 0; e < 16; ++e) {
            const int Kb = (e >> 3) * 16 + h * 8 + (e & 7);
            b[e] = vh[(k0 + Kb) * 80 + colB + mlo];
        }
        c = __builtin_amdgcn_wmma_f32_16x16x32_f16(
                /*neg_a=*/false, a, /*neg_b=*/false, b,
                /*c_mod=*/(short)0, c, /*reuse_a=*/false, /*reuse_b=*/false);
    }
    const int mb = h * 8;
#pragma unroll
    for (int r = 0; r < 8; ++r)
        vmp[(rowB + mb + r) * 80 + colB + mlo] = c[r];
}

// ---------------------------------------------------------------------------
// Kernel 3: first CG product + wn mixing (depends only on x):
//   mixed[x][ROWB[L]+M][d] = sum_{chunk,m,n,c,e} vmp[l1][x,m,c] vmp[l2][x,n,e]
//                            * C[m,n,M] * wn_L[k*64 + c*8 + e, d]
// One thread per (x,d). Only the chunk loop is unrolled (keeps mix[] register-
// indexed); inner loops rolled, operands read from L2-resident global memory.
// ---------------------------------------------------------------------------
__global__ void __launch_bounds__(64)
mix_kernel(const float* __restrict__ vmp,
           const float* __restrict__ wn0,
           const float* __restrict__ wn1,
           const float* __restrict__ wn2,
           const float* __restrict__ cg,
           float* __restrict__ mixed) {
    int tid = blockIdx.x * blockDim.x + threadIdx.x;
    if (tid >= NNODES * TAU) return;
    const int x = tid >> 3, d = tid & 7;
    const float* vx = vmp + x * 80;

    float mix[9] = {0.f,0.f,0.f,0.f,0.f,0.f,0.f,0.f,0.f};

#pragma unroll
    for (int ci = 0; ci < 15; ++ci) {
        const int l1 = CH_L1[ci], l2 = CH_L2[ci], L = CH_L[ci];
        const int n1 = 2*l1+1, n2 = 2*l2+1, nM = 2*L+1;
        const float* wn = (L == 0) ? wn0 : ((L == 1) ? wn1 : wn2);
        const float* wnb = wn + CH_K[ci] * 64 * 8 + d;
        for (int m = 0; m < n1; ++m) {
            for (int n = 0; n < n2; ++n) {
                float P = 0.f;
                for (int cc = 0; cc < 8; ++cc) {
                    const float av = vx[LBASE[l1] + m*8 + cc];
                    for (int e = 0; e < 8; ++e)
                        P += av * vx[LBASE[l2] + n*8 + e] * wnb[(cc*8 + e) * 8];
                }
                const float* cgm = cg + CH_OFF[ci] + (m*n2 + n)*nM;
#pragma unroll
                for (int M = 0; M < nM; ++M)
                    mix[ROWB[L] + M] += cgm[M] * P;
            }
        }
    }
    for (int r = 0; r < 9; ++r)
        mixed[x * 72 + r * 8 + d] = mix[r];
}

// ---------------------------------------------------------------------------
// Kernel 4: bandwidth-dominant phase. Block = one y, thread = one x.
// Per cell (x,y): T[m][M] = sum_n C[m,n,M]*s_l2[n];
//                 cgv[M]  = sum_m mixed[l1][x][m][cin]*T[m][M];
//                 acc[ROWB[L]+M][d] += cgv[M] * wr_L[x,y, k*8+cin, d]
// wr streamed with non-temporal 128-bit loads (252 MB read-once > 192 MB L2).
// Chunk/m/M/n loops unrolled so acc[72], T, sv stay in VGPRs; cin loop rolled.
// Deterministic LDS tree reduction over x, single write per (y, element).
// ---------------------------------------------------------------------------
__global__ void __launch_bounds__(NNODES)
cell_kernel(const float* __restrict__ mixed,
            const float* __restrict__ s0,
            const float* __restrict__ s1,
            const float* __restrict__ s2,
            const float* __restrict__ wr0,
            const float* __restrict__ wr1,
            const float* __restrict__ wr2,
            const float* __restrict__ cg,
            float* __restrict__ vsum) {
    const int y = blockIdx.x;
    const int x = threadIdx.x;
    const int cell = x * NNODES + y;

    __shared__ float cgs[CG_TOTAL];
    __shared__ float red[NNODES * 8];
    for (int i = threadIdx.x; i < CG_TOTAL; i += blockDim.x) cgs[i] = cg[i];
    __syncthreads();

    // spherical-harmonic values: sv[0]=l0, sv[1..3]=l1, sv[4..8]=l2
    float sv[9];
    sv[0] = s0[cell];
#pragma unroll
    for (int n = 0; n < 3; ++n) sv[1 + n] = s1[(size_t)cell * 3 + n];
#pragma unroll
    for (int n = 0; n < 5; ++n) sv[4 + n] = s2[(size_t)cell * 5 + n];

    float acc[72];
#pragma unroll
    for (int i = 0; i < 72; ++i) acc[i] = 0.f;

#pragma unroll
    for (int ci = 0; ci < 15; ++ci) {
        const int l1 = CH_L1[ci], l2 = CH_L2[ci], L = CH_L[ci];
        const int n1 = 2*l1+1, n2 = 2*l2+1, nM = 2*L+1;
        const float* wrL = (L == 0) ? wr0 : ((L == 1) ? wr1 : wr2);
        const float* wrp = wrL + ((size_t)cell * CDIM[L] + CH_K[ci] * 8) * 8;
        const float* mrow = mixed + x * 72 + ROWB[l1] * 8;

        // T[m][M] = sum_n C[m,n,M] * s_l2[n]   (all indices compile-time)
        float T[25];
#pragma unroll
        for (int m = 0; m < n1; ++m) {
#pragma unroll
            for (int M = 0; M < nM; ++M) {
                float t = 0.f;
#pragma unroll
                for (int n = 0; n < n2; ++n)
                    t += cgs[CH_OFF[ci] + (m*n2 + n)*nM + M] * sv[ROWB[l2] + n];
                T[m * nM + M] = t;
            }
        }

        const int accb = ROWB[L] * 8;
        for (int cin = 0; cin < 8; ++cin) {   // rolled: runtime offsets OK
            v4f w0 = __builtin_nontemporal_load((const v4f*)(wrp + cin * 8));
            v4f w1 = __builtin_nontemporal_load((const v4f*)(wrp + cin * 8 + 4));
            float cgv[5];
#pragma unroll
            for (int M = 0; M < nM; ++M) {
                float c = 0.f;
#pragma unroll
                for (int m = 0; m < n1; ++m)
                    c += mrow[m * 8 + cin] * T[m * nM + M];
                cgv[M] = c;
            }
#pragma unroll
            for (int M = 0; M < nM; ++M) {
                const int a = accb + M * 8;
                acc[a + 0] += cgv[M] * w0[0];
                acc[a + 1] += cgv[M] * w0[1];
                acc[a + 2] += cgv[M] * w0[2];
                acc[a + 3] += cgv[M] * w0[3];
                acc[a + 4] += cgv[M] * w1[0];
                acc[a + 5] += cgv[M] * w1[1];
                acc[a + 6] += cgv[M] * w1[2];
                acc[a + 7] += cgv[M] * w1[3];
            }
        }
    }

    // deterministic tree reduction over x (9 groups of 8 elements)
    __syncthreads();
    for (int g = 0; g < 9; ++g) {
#pragma unroll
        for (int j = 0; j < 8; ++j) red[x * 8 + j] = acc[g * 8 + j];
        __syncthreads();
        for (int off = NNODES / 2; off > 0; off >>= 1) {
            if (x < off) {
#pragma unroll
                for (int j = 0; j < 8; ++j)
                    red[x * 8 + j] += red[(x + off) * 8 + j];
            }
            __syncthreads();
        }
        if (x < 8) vsum[y * 72 + g * 8 + x] = red[x];
        __syncthreads();
    }
}

// ---------------------------------------------------------------------------
// Kernel 5: per-part normalization (divide by global sum of each part);
// vsum layout already matches the concatenated output layout.
// ---------------------------------------------------------------------------
__global__ void __launch_bounds__(NNODES)
norm_kernel(const float* __restrict__ vsum, float* __restrict__ out) {
    __shared__ float r0[NNODES], r1[NNODES], r2[NNODES];
    const int y = threadIdx.x;
    float p0 = 0.f, p1 = 0.f, p2 = 0.f;
    for (int i = 0; i < 8; ++i)   p0 += vsum[y * 72 + i];
    for (int i = 8; i < 32; ++i)  p1 += vsum[y * 72 + i];
    for (int i = 32; i < 72; ++i) p2 += vsum[y * 72 + i];
    r0[y] = p0; r1[y] = p1; r2[y] = p2;
    __syncthreads();
    for (int off = NNODES / 2; off > 0; off >>= 1) {
        if (y < off) { r0[y] += r0[y+off]; r1[y] += r1[y+off]; r2[y] += r2[y+off]; }
        __syncthreads();
    }
    const float i0 = 1.f / r0[0], i1 = 1.f / r1[0], i2 = 1.f / r2[0];
    for (int i = 0; i < 8; ++i)   out[y * 72 + i] = vsum[y * 72 + i] * i0;
    for (int i = 8; i < 32; ++i)  out[y * 72 + i] = vsum[y * 72 + i] * i1;
    for (int i = 32; i < 72; ++i) out[y * 72 + i] = vsum[y * 72 + i] * i2;
}

// ---------------------------------------------------------------------------
extern "C" void kernel_launch(void* const* d_in, const int* in_sizes, int n_in,
                              void* d_out, int out_size, void* d_ws, size_t ws_size,
                              hipStream_t stream) {
    (void)in_sizes; (void)n_in; (void)out_size; (void)ws_size;
    const float* v0  = (const float*)d_in[0];
    const float* v1  = (const float*)d_in[1];
    const float* v2  = (const float*)d_in[2];
    const float* adj = (const float*)d_in[3];
    const float* s0  = (const float*)d_in[4];
    const float* s1  = (const float*)d_in[5];
    const float* s2  = (const float*)d_in[6];
    const float* wn0 = (const float*)d_in[7];
    const float* wn1 = (const float*)d_in[8];
    const float* wn2 = (const float*)d_in[9];
    const float* wr0 = (const float*)d_in[10];
    const float* wr1 = (const float*)d_in[11];
    const float* wr2 = (const float*)d_in[12];
    float* out = (float*)d_out;

    // workspace layout (float slots)
    float* ws      = (float*)d_ws;
    float* vmp     = ws;                        // 256*80 = 20480 f32
    float* mixed   = ws + 20480;                // 256*72 = 18432 f32
    float* vsum    = ws + 38912;                // 256*72 = 18432 f32
    float* cg      = ws + 57344;                // 615 f32 (pad to 1024)
    _Float16* adjh = (_Float16*)(ws + 58368);   // 65536 f16 = 32768 f32 slots
    _Float16* vh   = (_Float16*)(ws + 91136);   // 20480 f16 = 10240 f32 slots
    // total ~ 101376 f32 slots = ~405 KB

    cg_init_kernel<<<1, 128, 0, stream>>>(cg);
    pack_adj_kernel<<<(NNODES * NNODES + 255) / 256, 256, 0, stream>>>(adj, adjh);
    pack_v_kernel<<<(NNODES * 80 + 255) / 256, 256, 0, stream>>>(v0, v1, v2, vh);
    adj_gemm_wmma<<<dim3(5, 16), 32, 0, stream>>>(adjh, vh, vmp);
    mix_kernel<<<(NNODES * TAU + 63) / 64, 64, 0, stream>>>(vmp, wn0, wn1, wn2, cg, mixed);
    cell_kernel<<<NNODES, NNODES, 0, stream>>>(mixed, s0, s1, s2, wr0, wr1, wr2, cg, vsum);
    norm_kernel<<<1, NNODES, 0, stream>>>(vsum, out);
}
